// DeformableTransformerEncoderLayer_65051574665314
// MI455X (gfx1250) — matrix operand order repaired and checked by reading, not compiled, not run on hardware
//
#include <hip/hip_runtime.h>
#include <hip/hip_bf16.h>
#include <cstdint>
#include <cstddef>

#define EMBED 256
#define HEADS 8
#define LEVELS 4
#define POINTS 4
#define FFN_DIM 1024
#define HD 32

typedef __attribute__((ext_vector_type(16))) _Float16 v16h;
typedef __attribute__((ext_vector_type(8)))  float    v8f;
typedef __attribute__((ext_vector_type(4)))  float    f4raw;

union Frag16 { f4raw f4[2]; v16h h; };

// ---------------------------------------------------------------------------
// f32 -> f16 convert (same layout)
// ---------------------------------------------------------------------------
__global__ void cvt_f16_kernel(const float* __restrict__ in, _Float16* __restrict__ out, size_t n) {
  size_t i = (size_t)blockIdx.x * blockDim.x + threadIdx.x;
  if (i < n) out[i] = (_Float16)in[i];
}

// f32 (K x N row-major) -> f16 transposed (N x K row-major)
__global__ void cvt_T_f16_kernel(const float* __restrict__ W, _Float16* __restrict__ Wt, int K, int N) {
  size_t i = (size_t)blockIdx.x * blockDim.x + threadIdx.x;
  if (i < (size_t)K * N) {
    int k = (int)(i / N), n = (int)(i % N);
    Wt[(size_t)n * K + k] = (_Float16)W[i];
  }
}

// ---------------------------------------------------------------------------
// WMMA GEMM: C[M,N] = A[M,K](f16, row-major) * Bt[N,K](f16, row-major)^T
// EPI 0: outF = acc + bias
// EPI 1: outF = acc + bias + resid
// EPI 2: outH = (f16) relu(acc + bias)
// Block: 128 threads (4 waves), 64x64 output tile, 2x2 waves of 32x32 each.
// ---------------------------------------------------------------------------
template <int EPI>
__global__ __launch_bounds__(128)
void gemm_wmma_kernel(const _Float16* __restrict__ A, const _Float16* __restrict__ Bt,
                      const float* __restrict__ bias, const float* __restrict__ resid,
                      float* __restrict__ outF, _Float16* __restrict__ outH,
                      int M, int N, int K) {
  __shared__ _Float16 lA[64 * 32];
  __shared__ _Float16 lB[64 * 32];

  const int tid   = threadIdx.x;
  const int m0    = blockIdx.x * 64;
  const int n0    = blockIdx.y * 64;
  const int wave  = tid >> 5;
  const int lane  = tid & 31;
  const int wm    = (wave >> 1) * 32;     // wave row offset in tile
  const int wn    = (wave & 1) * 32;      // wave col offset in tile
  const int laneM = lane & 15;
  const int hi    = lane >> 4;            // half-wave select

  const int rowL = tid >> 1;              // 0..63 staging row
  const int seg  = tid & 1;               // which 16-half segment

  const v8f vzero = {0.f, 0.f, 0.f, 0.f, 0.f, 0.f, 0.f, 0.f};
  v8f acc[2][2];
  acc[0][0] = vzero; acc[0][1] = vzero; acc[1][0] = vzero; acc[1][1] = vzero;

  for (int k0 = 0; k0 < K; k0 += 32) {
    // ---- stage A tile (64 x 32 halves) ----
    f4raw a0 = {0.f, 0.f, 0.f, 0.f}, a1 = a0;
    const int gr = m0 + rowL;
    if (gr < M) {
      const f4raw* p = (const f4raw*)(A + (size_t)gr * K + k0 + seg * 16);
      a0 = p[0]; a1 = p[1];
      if (k0 + 32 < K) __builtin_prefetch(A + (size_t)gr * K + k0 + 32, 0, 0);
    }
    // ---- stage B tile (Bt rows n0..n0+63, 32 K-halves) ----
    const f4raw* pb = (const f4raw*)(Bt + (size_t)(n0 + rowL) * K + k0 + seg * 16);
    f4raw b0 = pb[0], b1 = pb[1];

    f4raw* qa = (f4raw*)&lA[rowL * 32 + seg * 16];
    qa[0] = a0; qa[1] = a1;
    f4raw* qb = (f4raw*)&lB[rowL * 32 + seg * 16];
    qb[0] = b0; qb[1] = b1;
    __syncthreads();

    // ---- fragment loads (documented wave32 layouts) ----
    // A: lane m = laneM, element e -> K = (e&7) + 8*hi + 16*(e>>3)
    Frag16 fa[2], fb[2];
#pragma unroll
    for (int i = 0; i < 2; i++) {
      const _Float16* pr = &lA[(wm + 16 * i + laneM) * 32];
      fa[i].f4[0] = *(const f4raw*)&pr[8 * hi];
      fa[i].f4[1] = *(const f4raw*)&pr[16 + 8 * hi];
    }
    // B: lane n = laneM, element e -> K = (e&7) + 8*hi + 16*(e>>3)
#pragma unroll
    for (int j = 0; j < 2; j++) {
      const _Float16* pr = &lB[(wn + 16 * j + laneM) * 32];
      fb[j].f4[0] = *(const f4raw*)&pr[8 * hi];
      fb[j].f4[1] = *(const f4raw*)&pr[16 + 8 * hi];
    }
#pragma unroll
    for (int i = 0; i < 2; i++)
#pragma unroll
      for (int j = 0; j < 2; j++)
        acc[i][j] = __builtin_amdgcn_wmma_f32_16x16x32_f16(
            false, fa[i].h, false, fb[j].h, (short)0, acc[i][j], false, false);
    __syncthreads();
  }

  // ---- epilogue: D layout lane n = laneM, VGPR r -> M = r + 8*hi ----
#pragma unroll
  for (int i = 0; i < 2; i++) {
#pragma unroll
    for (int j = 0; j < 2; j++) {
      const int col = n0 + wn + 16 * j + laneM;
#pragma unroll
      for (int r = 0; r < 8; r++) {
        const int row = m0 + wm + 16 * i + r + 8 * hi;
        if (row < M) {
          float v = acc[i][j][r] + bias[col];
          const size_t idx = (size_t)row * N + col;
          if (EPI == 0)       outF[idx] = v;
          else if (EPI == 1)  outF[idx] = v + resid[idx];
          else                outH[idx] = (_Float16)fmaxf(v, 0.f);
        }
      }
    }
  }
}

// ---------------------------------------------------------------------------
// Per-(token, head): softmax over 16 attn logits (in place) and
// sampling loc = ref + off / [W,H] (in place over off buffer).
// ---------------------------------------------------------------------------
__global__ void softmax_loc_kernel(float* __restrict__ att, float* __restrict__ off,
                                   const float* __restrict__ ref, const int* __restrict__ shp,
                                   int M) {
  int t = blockIdx.x * blockDim.x + threadIdx.x;
  if (t >= M * HEADS) return;
  const int token = t >> 3, h = t & 7;

  float* lg = att + (size_t)token * (HEADS * LEVELS * POINTS) + h * (LEVELS * POINTS);
  float mx = lg[0];
  for (int i = 1; i < 16; i++) mx = fmaxf(mx, lg[i]);
  float e[16], s = 0.f;
  for (int i = 0; i < 16; i++) { e[i] = __expf(lg[i] - mx); s += e[i]; }
  const float inv = 1.f / s;
  for (int i = 0; i < 16; i++) lg[i] = e[i] * inv;

  float* o = off + (size_t)token * EMBED + h * (LEVELS * POINTS * 2);
  const float* rp = ref + (size_t)token * (LEVELS * 2);
  for (int l = 0; l < LEVELS; l++) {
    const float Wf = (float)shp[2 * l + 1], Hf = (float)shp[2 * l];
    const float rx = rp[2 * l], ry = rp[2 * l + 1];
    for (int p = 0; p < POINTS; p++) {
      const int k = (l * POINTS + p) * 2;
      o[k]     = rx + o[k]     / Wf;
      o[k + 1] = ry + o[k + 1] / Hf;
    }
  }
}

// ---------------------------------------------------------------------------
// Deformable sampling core: one wave per (token, head), lane = channel (HD=32).
// value buffer (~50MB f32) fits in 192MB L2 -> gathers served by L2.
// ---------------------------------------------------------------------------
__global__ __launch_bounds__(256)
void deform_core_kernel(const float* __restrict__ value, const float* __restrict__ loc,
                        const float* __restrict__ att, const int* __restrict__ shp,
                        const int* __restrict__ lsi, _Float16* __restrict__ out,
                        int M, int nq) {
  const int wave = (blockIdx.x * blockDim.x + threadIdx.x) >> 5;
  if (wave >= M * HEADS) return;
  const int lane  = threadIdx.x & 31;
  const int token = wave >> 3, h = wave & 7;
  const int b     = token / nq;

  const float* lc = loc + (size_t)token * EMBED + h * (LEVELS * POINTS * 2);
  const float* aw = att + (size_t)token * (HEADS * LEVELS * POINTS) + h * (LEVELS * POINTS);

  float acc = 0.f;
  for (int l = 0; l < LEVELS; l++) {
    const int Hl = shp[2 * l], Wl = shp[2 * l + 1];
    const int st = lsi[l];
    const size_t vbase = ((size_t)(b * nq + st)) * EMBED + h * HD + lane;
    for (int p = 0; p < POINTS; p++) {
      const float lx = lc[(l * POINTS + p) * 2];
      const float ly = lc[(l * POINTS + p) * 2 + 1];
      const float a  = aw[l * POINTS + p];
      const float x = lx * (float)Wl - 0.5f;
      const float y = ly * (float)Hl - 0.5f;
      const float x0f = floorf(x), y0f = floorf(y);
      const float fx = x - x0f, fy = y - y0f;
      const int x0 = (int)x0f, y0 = (int)y0f;
#pragma unroll
      for (int dy = 0; dy < 2; dy++)
#pragma unroll
        for (int dx = 0; dx < 2; dx++) {
          const int xi = x0 + dx, yi = y0 + dy;
          float w = (dx ? fx : 1.f - fx) * (dy ? fy : 1.f - fy);
          const bool valid = (xi >= 0) && (xi < Wl) && (yi >= 0) && (yi < Hl);
          const int xc = min(max(xi, 0), Wl - 1);
          const int yc = min(max(yi, 0), Hl - 1);
          const float wv = valid ? (a * w) : 0.f;
          acc += wv * value[vbase + (size_t)(yc * Wl + xc) * EMBED];
        }
    }
  }
  out[(size_t)token * EMBED + h * HD + lane] = (_Float16)acc;
}

// ---------------------------------------------------------------------------
// LayerNorm over 256 channels, one wave32 per token, shfl_xor reductions.
// Writes f32 (residual path) and optionally f16 (next GEMM input).
// ---------------------------------------------------------------------------
__global__ __launch_bounds__(256)
void layernorm_kernel(const float* __restrict__ in, const float* __restrict__ g,
                      const float* __restrict__ bt, float* __restrict__ outF,
                      _Float16* __restrict__ outH, int M) {
  const int wave = (blockIdx.x * blockDim.x + threadIdx.x) >> 5;
  if (wave >= M) return;
  const int lane = threadIdx.x & 31;
  const float* x = in + (size_t)wave * EMBED;

  float v[8], s = 0.f;
#pragma unroll
  for (int i = 0; i < 8; i++) { v[i] = x[lane + 32 * i]; s += v[i]; }
#pragma unroll
  for (int o = 16; o >= 1; o >>= 1) s += __shfl_xor(s, o, 32);
  const float mean = s * (1.f / EMBED);

  float s2 = 0.f;
#pragma unroll
  for (int i = 0; i < 8; i++) { const float d = v[i] - mean; s2 += d * d; }
#pragma unroll
  for (int o = 16; o >= 1; o >>= 1) s2 += __shfl_xor(s2, o, 32);
  const float rinv = rsqrtf(s2 * (1.f / EMBED) + 1e-5f);

#pragma unroll
  for (int i = 0; i < 8; i++) {
    const int col = lane + 32 * i;
    const float r = (v[i] - mean) * rinv * g[col] + bt[col];
    outF[(size_t)wave * EMBED + col] = r;
    if (outH) outH[(size_t)wave * EMBED + col] = (_Float16)r;
  }
}

// ---------------------------------------------------------------------------
extern "C" void kernel_launch(void* const* d_in, const int* in_sizes, int n_in,
                              void* d_out, int out_size, void* d_ws, size_t ws_size,
                              hipStream_t stream) {
  const float* src   = (const float*)d_in[0];
  const float* ref   = (const float*)d_in[1];
  const int*   shp   = (const int*)d_in[2];
  const int*   lsi   = (const int*)d_in[3];
  const float* w_off = (const float*)d_in[4];
  const float* b_off = (const float*)d_in[5];
  const float* w_att = (const float*)d_in[6];
  const float* b_att = (const float*)d_in[7];
  const float* w_val = (const float*)d_in[8];
  const float* b_val = (const float*)d_in[9];
  const float* w_out = (const float*)d_in[10];
  const float* b_out = (const float*)d_in[11];
  const float* g1    = (const float*)d_in[12];
  const float* bt1   = (const float*)d_in[13];
  const float* w1    = (const float*)d_in[14];
  const float* b1    = (const float*)d_in[15];
  const float* w2    = (const float*)d_in[16];
  const float* b2    = (const float*)d_in[17];
  const float* g2    = (const float*)d_in[18];
  const float* bt2   = (const float*)d_in[19];

  const int M  = in_sizes[0] / EMBED;   // B * nq tokens
  const int nq = M / 2;                 // reference uses B == 2

  // ---- workspace bump allocator (256B aligned) ----
  char* ws = (char*)d_ws;
  size_t cur = 0;
  auto alloc = [&](size_t bytes) -> char* {
    cur = (cur + 255) & ~(size_t)255;
    char* p = ws + cur;
    cur += bytes;
    return p;
  };

  _Float16* srcH = (_Float16*)alloc((size_t)M * EMBED * 2);
  _Float16* wtv  = (_Float16*)alloc((size_t)256 * 256 * 2);
  _Float16* wto  = (_Float16*)alloc((size_t)256 * 256 * 2);
  _Float16* wta  = (_Float16*)alloc((size_t)128 * 256 * 2);
  _Float16* wtou = (_Float16*)alloc((size_t)256 * 256 * 2);
  _Float16* w1t  = (_Float16*)alloc((size_t)FFN_DIM * 256 * 2);
  _Float16* w2t  = (_Float16*)alloc((size_t)256 * FFN_DIM * 2);
  float*    val  = (float*)alloc((size_t)M * EMBED * 4);
  float*    offp = (float*)alloc((size_t)M * EMBED * 4);
  float*    att  = (float*)alloc((size_t)M * 128 * 4);
  _Float16* h1   = (_Float16*)alloc((size_t)M * FFN_DIM * 2);

  // lifetime-based aliases
  _Float16* coreH = srcH;           // after the three src-projections finish
  float*    xpre  = val;            // after deform core finishes
  float*    xbuf  = offp;           // loc dead after core
  _Float16* xh    = (_Float16*)att; // attn dead after core (same byte size)
  float*    ypre  = val;            // xpre dead after LN1

  const dim3 blk256(256);
  const dim3 gemmBlk(128);
  const int  mtiles = (M + 63) / 64;

  // 1. convert src -> f16
  {
    size_t n = (size_t)M * EMBED;
    cvt_f16_kernel<<<dim3((unsigned)((n + 255) / 256)), blk256, 0, stream>>>(src, srcH, n);
  }
  // 2. transpose+convert weights (one-time, tiny)
  cvt_T_f16_kernel<<<dim3((256 * 256 + 255) / 256), blk256, 0, stream>>>(w_val, wtv, 256, 256);
  cvt_T_f16_kernel<<<dim3((256 * 256 + 255) / 256), blk256, 0, stream>>>(w_off, wto, 256, 256);
  cvt_T_f16_kernel<<<dim3((256 * 128 + 255) / 256), blk256, 0, stream>>>(w_att, wta, 256, 128);
  cvt_T_f16_kernel<<<dim3((256 * 256 + 255) / 256), blk256, 0, stream>>>(w_out, wtou, 256, 256);
  cvt_T_f16_kernel<<<dim3((256 * FFN_DIM + 255) / 256), blk256, 0, stream>>>(w1, w1t, 256, FFN_DIM);
  cvt_T_f16_kernel<<<dim3((FFN_DIM * 256 + 255) / 256), blk256, 0, stream>>>(w2, w2t, FFN_DIM, 256);

  // 3. projections (WMMA GEMMs)
  gemm_wmma_kernel<0><<<dim3(mtiles, 4), gemmBlk, 0, stream>>>(srcH, wtv, b_val, nullptr, val, nullptr, M, 256, 256);
  gemm_wmma_kernel<0><<<dim3(mtiles, 4), gemmBlk, 0, stream>>>(srcH, wto, b_off, nullptr, offp, nullptr, M, 256, 256);
  gemm_wmma_kernel<0><<<dim3(mtiles, 2), gemmBlk, 0, stream>>>(srcH, wta, b_att, nullptr, att, nullptr, M, 128, 256);

  // 4. softmax + sampling locations (in place)
  softmax_loc_kernel<<<dim3((M * HEADS + 255) / 256), blk256, 0, stream>>>(att, offp, ref, shp, M);

  // 5. deformable sampling core -> f16 (one wave per token*head => M blocks of 8 waves)
  deform_core_kernel<<<dim3(M), blk256, 0, stream>>>(val, offp, att, shp, lsi, coreH, M, nq);

  // 6. output projection + residual(src) -> xpre
  gemm_wmma_kernel<1><<<dim3(mtiles, 4), gemmBlk, 0, stream>>>(coreH, wtou, b_out, src, xpre, nullptr, M, 256, 256);

  // 7. LN1 -> x (f32) + x (f16)
  layernorm_kernel<<<dim3((M + 7) / 8), blk256, 0, stream>>>(xpre, g1, bt1, xbuf, xh, M);

  // 8. FFN1: relu(x @ w1 + b1) -> h1 (f16)
  gemm_wmma_kernel<2><<<dim3(mtiles, FFN_DIM / 64), gemmBlk, 0, stream>>>(xh, w1t, b1, nullptr, nullptr, h1, M, FFN_DIM, 256);

  // 9. FFN2 + residual(x) -> ypre
  gemm_wmma_kernel<1><<<dim3(mtiles, 4), gemmBlk, 0, stream>>>(h1, w2t, b2, xbuf, ypre, nullptr, M, 256, FFN_DIM);

  // 10. LN2 -> d_out (f32)
  layernorm_kernel<<<dim3((M + 7) / 8), blk256, 0, stream>>>(ypre, g2, bt2, (float*)d_out, nullptr, M);

  (void)n_in; (void)out_size; (void)ws_size;
}
